// InvariantPolynomial_89850715832863
// MI455X (gfx1250) — matrix operand order repaired and checked by reading, not compiled, not run on hardware
//
#include <hip/hip_runtime.h>
#include <math.h>

#define N_NODES  20000
#define N_EDGES  320000
#define N_GRAPHS 256
#define NA 23
#define NB 7
#define M0 64
#define M1 24
#define M2 16

#define KTOT (NA*NB)      // 161
#define KP   192          // padded K: 6 steps of 32
#define KS   6            // K steps
#define NBLK 7            // N blocks of 16 (104 -> 112)
#define NOUT (M0+M1+M2)   // 104

typedef __attribute__((ext_vector_type(16))) _Float16 v16h;
typedef __attribute__((ext_vector_type(8)))  _Float16 v8h;
typedef __attribute__((ext_vector_type(8)))  float    v8f;

// workspace layout (bytes)
#define WC_OFF 0                                   // 7*6*32*16 halves = 43008 B
#define N0_OFF 65536
#define N1_OFF (N0_OFF + N_NODES*M0*4)             // n0: 5,120,000 B
#define N2_OFF (N1_OFF + N_NODES*M1*3*4)           // n1: 5,760,000 B
#define NO_OFF (N2_OFF + N_NODES*M2*5*4)           // n2: 6,400,000 B
// node_out: 80,000 B ; total ~17.4 MB

__global__ void zero_f32(float* __restrict__ p, int n) {
  int i = blockIdx.x * blockDim.x + threadIdx.x;
  int stride = gridDim.x * blockDim.x;
  for (; i < n; i += stride) p[i] = 0.0f;
}

// Build fused weight matrix W[k=u*7+v][n] (n: 0..63 -> W1, 64..87 -> W2, 88..103 -> W3)
// directly in WMMA B-fragment order:
//   frag index = ((nb*KS + ks)*32 + lane)*16 + j
//   column n = nb*16 + lane%16 ; K = ks*32 + (lane<16 ? 0 : 8) + (j<8 ? j : j+8)
__global__ void prep_wc(const float* __restrict__ W1, const float* __restrict__ W2,
                        const float* __restrict__ W3, _Float16* __restrict__ wc) {
  int idx = blockIdx.x * blockDim.x + threadIdx.x;
  const int total = NBLK * KS * 32 * 16;
  if (idx >= total) return;
  int j    = idx & 15;
  int lane = (idx >> 4) & 31;
  int ks   = (idx >> 9) % KS;
  int nb   = idx / (16 * 32 * KS);
  int n = nb * 16 + (lane & 15);
  int k = ks * 32 + ((lane < 16) ? 0 : 8) + ((j < 8) ? j : j + 8);
  float v = 0.0f;
  if (k < KTOT && n < NOUT) {
    if (n < M0)            v = W1[k * M0 + n];
    else if (n < M0 + M1)  v = W2[k * M1 + (n - M0)];
    else                   v = W3[k * M2 + (n - M0 - M1)];
  }
  wc[idx] = (_Float16)v;
}

// Phase 1: per 16-edge tile, build outer-product features in LDS, run
// [16 x 192] x [192 x 112] GEMM via v_wmma_f32_16x16x32_f16, then scatter
// sh-weighted results into per-node accumulators with atomics.
__global__ __launch_bounds__(256) void edge_phase1(
    const float* __restrict__ pos, const float* __restrict__ x,
    const float* __restrict__ ea,  const int*   __restrict__ eidx,
    const _Float16* __restrict__ wc,
    float* __restrict__ n0, float* __restrict__ n1, float* __restrict__ n2)
{
  __shared__ _Float16 As[8][16][KP];   // 49152 B
  __shared__ float sh1s[8][16][3];
  __shared__ float sh2s[8][16][5];
  __shared__ int   dsts[8][16];

  const int lane = threadIdx.x & 31;
  const int wid  = threadIdx.x >> 5;
  const int tile = blockIdx.x * 8 + wid;   // grid = 2500 -> tiles 0..19999
  const int tb   = tile * 16;

  const int row = lane & 15;
  const int e   = tb + row;
  const int src = eidx[e];
  const int dst = eidx[N_EDGES + e];

  const float s3  = 1.7320508075688772f;
  const float s5  = 2.2360679774997896f;
  const float s15 = 3.8729833462074170f;

  float px = pos[src*3+0] - pos[dst*3+0];
  float py = pos[src*3+1] - pos[dst*3+1];
  float pz = pos[src*3+2] - pos[dst*3+2];

  float eav[NB];
  #pragma unroll
  for (int v = 0; v < NB; v++) eav[v] = ea[e*NB + v];

  if (lane < 16) {
    dsts[wid][row] = dst;
    sh1s[wid][row][0] = s3*px; sh1s[wid][row][1] = s3*py; sh1s[wid][row][2] = s3*pz;
    float r2 = px*px + py*py + pz*pz;
    sh2s[wid][row][0] = s15*px*py;
    sh2s[wid][row][1] = s15*py*pz;
    sh2s[wid][row][2] = 0.5f*s5*(3.0f*pz*pz - r2);
    sh2s[wid][row][3] = s15*px*pz;
    sh2s[wid][row][4] = 0.5f*s15*(px*px - py*py);
  }

  // outer product: lane<16 covers u=0..11 (k 0..83), lane>=16 covers u=12..22 + zero pad
  int u0 = (lane < 16) ? 0  : 12;
  int u1 = (lane < 16) ? 12 : NA;
  for (int u = u0; u < u1; u++) {
    float xu = x[src*NA + u];
    #pragma unroll
    for (int v = 0; v < NB; v++)
      As[wid][row][u*NB + v] = (_Float16)(xu * eav[v]);
  }
  if (lane >= 16) {
    for (int k = KTOT; k < KP; k++) As[wid][row][k] = (_Float16)0.0f;
  }
  __syncthreads();

  v8f zero = {};
  v8f acc[NBLK];
  #pragma unroll
  for (int nb = 0; nb < NBLK; nb++) acc[nb] = zero;

  const int hl = (lane < 16) ? 0 : 8;
  #pragma unroll
  for (int ks = 0; ks < KS; ks++) {
    const _Float16* arow = &As[wid][row][ks*32 + hl];
    v8h alo = *(const v8h*)(arow);
    v8h ahi = *(const v8h*)(arow + 16);
    v16h a = __builtin_shufflevector(alo, ahi, 0,1,2,3,4,5,6,7,8,9,10,11,12,13,14,15);
    #pragma unroll
    for (int nb = 0; nb < NBLK; nb++) {
      v16h b = *(const v16h*)(wc + ((nb*KS + ks)*32 + lane)*16);
      acc[nb] = __builtin_amdgcn_wmma_f32_16x16x32_f16(
          false, a, false, b, (short)0, acc[nb], false, false);
    }
  }

  // C layout: element r of acc -> edge row (r + (lane<16?0:8)), channel nb*16 + lane%16
  const float a1 = 1.0f / sqrtf((float)(NA*NB));
  #pragma unroll
  for (int nb = 0; nb < NBLK; nb++) {
    int n = nb*16 + (lane & 15);
    #pragma unroll
    for (int r = 0; r < 8; r++) {
      int er = r + ((lane < 16) ? 0 : 8);
      float val = a1 * acc[nb][r];
      int d = dsts[wid][er];
      if (n < M0) {
        atomicAdd(&n0[d*M0 + n], val);
      } else if (n < M0 + M1) {
        int w = n - M0;
        #pragma unroll
        for (int mm = 0; mm < 3; mm++)
          atomicAdd(&n1[(d*M1 + w)*3 + mm], val * sh1s[wid][er][mm]);
      } else if (n < NOUT) {
        int w = n - M0 - M1;
        #pragma unroll
        for (int mm = 0; mm < 5; mm++)
          atomicAdd(&n2[(d*M2 + w)*5 + mm], val * sh2s[wid][er][mm]);
      }
    }
  }
}

// Phase 2: g[e] = c1*<n0[src], ea·V1> + c2*<n1[src]·sh1, ea·V2> + c3*<n2[src]·sh2, ea·V3>
// then atomic scatter into node_out[dst].
__global__ void edge_phase2(const float* __restrict__ pos, const float* __restrict__ ea,
                            const int* __restrict__ eidx,
                            const float* __restrict__ V1, const float* __restrict__ V2,
                            const float* __restrict__ V3,
                            const float* __restrict__ n0, const float* __restrict__ n1,
                            const float* __restrict__ n2, float* __restrict__ node_out)
{
  int e = blockIdx.x * blockDim.x + threadIdx.x;
  if (e >= N_EDGES) return;
  int src = eidx[e], dst = eidx[N_EDGES + e];

  const float s3  = 1.7320508075688772f;
  const float s5  = 2.2360679774997896f;
  const float s15 = 3.8729833462074170f;

  float px = pos[src*3+0] - pos[dst*3+0];
  float py = pos[src*3+1] - pos[dst*3+1];
  float pz = pos[src*3+2] - pos[dst*3+2];
  float r2 = px*px + py*py + pz*pz;
  float sh1[3] = { s3*px, s3*py, s3*pz };
  float sh2[5] = { s15*px*py, s15*py*pz, 0.5f*s5*(3.0f*pz*pz - r2),
                   s15*px*pz, 0.5f*s15*(px*px - py*py) };

  float eav[NB];
  #pragma unroll
  for (int v = 0; v < NB; v++) eav[v] = ea[e*NB + v];

  float g1 = 0.0f;
  for (int u = 0; u < M0; u++) {
    float p = 0.0f;
    #pragma unroll
    for (int v = 0; v < NB; v++) p += eav[v] * V1[u*NB + v];
    g1 += n0[src*M0 + u] * p;
  }
  float g2 = 0.0f;
  for (int u = 0; u < M1; u++) {
    float p = 0.0f;
    #pragma unroll
    for (int v = 0; v < NB; v++) p += eav[v] * V2[u*NB + v];
    float s = 0.0f;
    #pragma unroll
    for (int mm = 0; mm < 3; mm++) s += n1[(src*M1 + u)*3 + mm] * sh1[mm];
    g2 += p * s;
  }
  float g3 = 0.0f;
  for (int u = 0; u < M2; u++) {
    float p = 0.0f;
    #pragma unroll
    for (int v = 0; v < NB; v++) p += eav[v] * V3[u*NB + v];
    float s = 0.0f;
    #pragma unroll
    for (int mm = 0; mm < 5; mm++) s += n2[(src*M2 + u)*5 + mm] * sh2[mm];
    g3 += p * s;
  }

  float g = g1 * (1.0f/sqrtf(448.0f))
          + g2 * (1.0f/sqrtf(504.0f))
          + g3 * (1.0f/sqrtf(560.0f));
  atomicAdd(&node_out[dst], g);
}

__global__ void node_reduce(const int* __restrict__ batch,
                            const float* __restrict__ node_out,
                            float* __restrict__ out) {
  int n = blockIdx.x * blockDim.x + threadIdx.x;
  if (n >= N_NODES) return;
  atomicAdd(&out[batch[n]], node_out[n]);
}

extern "C" void kernel_launch(void* const* d_in, const int* in_sizes, int n_in,
                              void* d_out, int out_size, void* d_ws, size_t ws_size,
                              hipStream_t stream) {
  const float* pos = (const float*)d_in[0];
  const float* x   = (const float*)d_in[1];
  const float* ea  = (const float*)d_in[2];
  const int*   eidx  = (const int*)d_in[3];
  const int*   batch = (const int*)d_in[4];
  const float* W1 = (const float*)d_in[5];
  const float* W2 = (const float*)d_in[6];
  const float* W3 = (const float*)d_in[7];
  const float* V1 = (const float*)d_in[8];
  const float* V2 = (const float*)d_in[9];
  const float* V3 = (const float*)d_in[10];
  float* out = (float*)d_out;

  char* ws = (char*)d_ws;
  _Float16* wc = (_Float16*)(ws + WC_OFF);
  float* n0       = (float*)(ws + N0_OFF);
  float* n1       = (float*)(ws + N1_OFF);
  float* n2       = (float*)(ws + N2_OFF);
  float* node_out = (float*)(ws + NO_OFF);

  // zero accumulators (n0,n1,n2,node_out are contiguous) and output
  int zn = N_NODES*(M0 + M1*3 + M2*5) + N_NODES;   // 4,340,000 floats
  zero_f32<<<4096, 256, 0, stream>>>(n0, zn);
  zero_f32<<<1, 256, 0, stream>>>(out, N_GRAPHS);

  prep_wc<<<(NBLK*KS*32*16 + 255)/256, 256, 0, stream>>>(W1, W2, W3, wc);

  edge_phase1<<<N_EDGES/(16*8), 256, 0, stream>>>(pos, x, ea, eidx, wc, n0, n1, n2);
  edge_phase2<<<(N_EDGES + 255)/256, 256, 0, stream>>>(pos, ea, eidx, V1, V2, V3,
                                                       n0, n1, n2, node_out);
  node_reduce<<<(N_NODES + 255)/256, 256, 0, stream>>>(batch, node_out, out);
}
